// harmolearn_37821482009415
// MI455X (gfx1250) — matrix operand order repaired and checked by reading, not compiled, not run on hardware
//
#include <hip/hip_runtime.h>

// ---------------------------------------------------------------------------
// Problem constants (match the JAX reference)
// ---------------------------------------------------------------------------
#define CH    224      // channels (= T as well: inp is (B, 224, 224))
#define WIN   16       // correlation window length
#define NT    14       // 224 / 16 tiles per matrix edge
#define NWAVES 8       // 256-thread block = 8 wave32 on gfx1250

typedef _Float16 v16h __attribute__((ext_vector_type(16)));
typedef _Float16 v8h  __attribute__((ext_vector_type(8)));
typedef float    v8f  __attribute__((ext_vector_type(8)));

// ---------------------------------------------------------------------------
// Phase 1: one workgroup per (batch b, time t).
// Computes ccv[c] = minmax-normalized mean |corr| of the 224x16 window and
// stores it to ccs[b][t][c] in workspace.  The 224x224 Gram matrix is done
// with v_wmma_f32_16x16x32_f16 (K=16 real, K=16..31 zero-padded).
//
// Work split: each wave owns whole 16-row bands (tm) of the cov matrix and
// sweeps tn = 0..13.  A-operand and rd[M] are loaded once per band; the
// per-row |cc| sums are accumulated in registers across the sweep and flushed
// with 8 ds_add_f32 per band (instead of 8 per tile).
// ---------------------------------------------------------------------------
__global__ __launch_bounds__(256) void corr_phase1(
    const float* __restrict__ inp,   // (B, CH, CH)
    float*       __restrict__ ccs)   // (B, CH(t), CH(c))
{
    __shared__ __align__(32) _Float16 bc[CH * WIN]; // centered window, f16
    __shared__ float rd[CH];                        // 1/sqrt(diag(cov))
    __shared__ float vsum[CH];                      // row sums of |cc|
    __shared__ float norm[2];                       // {min, 1/(max-min)}

    const int t   = blockIdx.x;   // time step 0..223
    const int b   = blockIdx.y;   // batch      0..B-1
    const int tid = threadIdx.x;

    // --- Step 1: per-channel window load, mean-center, f16 convert, diag ---
    if (tid < CH) {
        const float* row = inp + ((size_t)b * CH + tid) * CH;
        float v[WIN];
        float s = 0.f;
        #pragma unroll
        for (int j = 0; j < WIN; ++j) {
            const int src = t - WIN + j;            // left zero-padding
            const float x = (src >= 0) ? row[src] : 0.f;
            v[j] = x;
            s += x;
        }
        const float mean = s * (1.0f / WIN);
        float ss = 0.f;
        #pragma unroll
        for (int j = 0; j < WIN; ++j) {
            const float cv = v[j] - mean;
            ss += cv * cv;
            bc[tid * WIN + j] = (_Float16)cv;
        }
        rd[tid]   = rsqrtf(ss);                     // 1/d_i
        vsum[tid] = 0.f;
    }
    __syncthreads();

    // --- Step 2: cov bands via WMMA, row sums folded in registers ----------
    const int wave = tid >> 5;
    const int lane = tid & 31;
    const int half = lane >> 4;   // 0: lanes 0-15, 1: lanes 16-31
    const int lm   = lane & 15;

    for (int tm = wave; tm < NT; tm += NWAVES) {
        // A operand (band-invariant): lane holds row (tm*16+lm),
        // K = 8*half .. 8*half+7 in regs 0..7; K=16..31 zero pad
        // (ISA 7.12.2 f16 A layout).
        v16h a = {};
        {
            const v8h av = *(const v8h*)&bc[(tm * 16 + lm) * WIN + half * 8];
            #pragma unroll
            for (int j = 0; j < 8; ++j) a[j] = av[j];
        }
        // rd for the 8 output rows this lane accumulates (band-invariant).
        float rdm[8];
        #pragma unroll
        for (int r = 0; r < 8; ++r) rdm[r] = rd[tm * 16 + r + 8 * half];

        float acc[8];
        #pragma unroll
        for (int r = 0; r < 8; ++r) acc[r] = 0.f;

        for (int tn = 0; tn < NT; ++tn) {
            // B operand (= Bc^T): lane holds column (tn*16+lm),
            // K = 8*half .. 8*half+7 in regs 0..7; K=16..31 zero pad.
            v16h bb = {};
            {
                const v8h bv = *(const v8h*)&bc[(tn * 16 + lm) * WIN + half * 8];
                #pragma unroll
                for (int j = 0; j < 8; ++j) bb[j] = bv[j];
            }

            v8f c = {};
            c = __builtin_amdgcn_wmma_f32_16x16x32_f16(
                    /*neg_a=*/false, a, /*neg_b=*/false, bb,
                    /*c_mod=*/(short)0, c, /*reuse_a=*/false, /*reuse_b=*/false);

            // D layout: reg r -> M = tm*16 + r + 8*half, N = tn*16 + lm.
            const float rdn = rd[tn * 16 + lm];
            #pragma unroll
            for (int r = 0; r < 8; ++r) {
                // clip(cov/(d_i d_j),-1,1)+abs == min(|cov|*rd_i*rd_j, 1)
                acc[r] += fminf(fabsf(c[r]) * rdm[r] * rdn, 1.0f);
            }
        }

        // Flush the lane-partial row sums (each row still needs the sum over
        // the 16 lanes that held its columns): 8 atomics per band.
        #pragma unroll
        for (int r = 0; r < 8; ++r) {
            atomicAdd(&vsum[tm * 16 + r + 8 * half], acc[r]);  // ds_add_f32
        }
    }
    __syncthreads();

    // --- Step 3: min-max normalize (scale/shift invariant: skip /224) ------
    if (wave == 0) {
        float mn =  3.4e38f, mx = -3.4e38f;
        for (int i = lane; i < CH; i += 32) {
            const float x = vsum[i];
            mn = fminf(mn, x);
            mx = fmaxf(mx, x);
        }
        #pragma unroll
        for (int off = 16; off > 0; off >>= 1) {
            mn = fminf(mn, __shfl_xor(mn, off, 32));
            mx = fmaxf(mx, __shfl_xor(mx, off, 32));
        }
        if (lane == 0) {
            norm[0] = mn;
            norm[1] = 1.0f / (mx - mn);
        }
    }
    __syncthreads();

    if (tid < CH) {
        ccs[((size_t)b * CH + t) * CH + tid] = (vsum[tid] - norm[0]) * norm[1];
    }
}

// ---------------------------------------------------------------------------
// Phase 2: per-(batch, channel) leaky-integrate-and-fire scan.
// w carries across batches in the reference, but w <- 0.1 w + 0.9 ccv decays
// 10x per step, so a 48-step lookback into the previous batch reproduces the
// carry to (beyond) f32 precision -> all batches run in parallel.
// ---------------------------------------------------------------------------
__global__ __launch_bounds__(CH) void snn_phase2(
    const float* __restrict__ inp,   // (B, CH, CH)
    const float* __restrict__ w0,    // (CH, 1)
    const float* __restrict__ ccs,   // (B, CH(t), CH(c))
    float*       __restrict__ out)   // (B, 1, CH(c), CH(t))
{
    const int b = blockIdx.x;
    const int c = threadIdx.x;       // 0..223

    float w;
    if (b == 0) {
        w = w0[c];
    } else {
        const float* ccp = ccs + (size_t)(b - 1) * CH * CH;
        w = 0.f;
        for (int t = CH - 48; t < CH; ++t)          // 0.1^48 ~ underflow
            w = 0.1f * w + 0.9f * ccp[(size_t)t * CH + c];
    }

    const float* ccb  = ccs + (size_t)b * CH * CH;
    const float* xr   = inp + ((size_t)b * CH + c) * CH;
    float*       orow = out + ((size_t)b * CH + c) * CH;

    float mem = 0.f;
    for (int t = 0; t < CH; ++t) {
        w = 0.1f * w + 0.9f * ccb[(size_t)t * CH + c];
        const float wc    = fminf(fmaxf(w, -1.f), 1.f);
        const float I     = wc * xr[t];
        const float reset = (mem > 1.0f) ? 1.0f : 0.0f;   // THR = 1
        mem = 0.95f * mem + I - reset;                    // BETA = 0.95
        orow[t] = (mem > 1.0f) ? 1.0f : 0.0f;
    }
}

// ---------------------------------------------------------------------------
extern "C" void kernel_launch(void* const* d_in, const int* in_sizes, int n_in,
                              void* d_out, int out_size, void* d_ws, size_t ws_size,
                              hipStream_t stream)
{
    const float* inp = (const float*)d_in[0];
    const float* w0  = (const float*)d_in[1];
    float*       out = (float*)d_out;
    float*       ccs = (float*)d_ws;          // needs B*CH*CH*4 bytes (~49 MB)

    const int B = in_sizes[0] / (CH * CH);    // 256

    dim3 g1(CH, B);                            // one block per (t, b)
    corr_phase1<<<g1, 256, 0, stream>>>(inp, ccs);
    snn_phase2<<<B, CH, 0, stream>>>(inp, w0, ccs, out);
}
